// PhysRNN_85383949845104
// MI455X (gfx1250) — compile-verified, matrix operand order checked by reading, not run on hardware
//
#include <hip/hip_runtime.h>
#include <hip/hip_bf16.h>

// ---------------------------------------------------------------------------
// PhysRNN for MI455X (gfx1250, wave32).
//  - bf16 WMMA (v_wmma_f32_16x16x32_bf16) for all GEMMs, f32 accumulate.
//  - A stream copied with GLOBAL_LOAD_ASYNC_TO_LDS_B128 (ASYNCcnt) when the
//    toolchain exposes the builtin; register-staged fallback otherwise.
//  - fully double-buffered LDS -> single barrier per K-chunk.
//  - 20 independent branch projections batched into one 1280-WG launch.
//  - sequential RNN scans: 1 WG of 512 threads, h in LDS, W_hh^T coalesced.
// ---------------------------------------------------------------------------

typedef __attribute__((ext_vector_type(16))) __bf16 v16bf;
typedef __attribute__((ext_vector_type(8)))  float  v8f;
typedef __attribute__((ext_vector_type(4)))  float  v4f;
typedef __attribute__((ext_vector_type(4)))  int    v4i;

#define N_HID   512
#define L_SEQ   512
#define F_TOTAL 311347

#if defined(__HIP_DEVICE_COMPILE__) && \
    __has_builtin(__builtin_amdgcn_global_load_async_to_lds_b128)
#define USE_ASYNC_LDS 1
#else
#define USE_ASYNC_LDS 0
#endif

#if USE_ASYNC_LDS
// builtin signature (from clang diagnostic): pointers to v4i in the global
// (AS1) and LDS (AS3) address spaces, then i32 offset + i32 cpol.
typedef __attribute__((address_space(1))) v4i* gptr_v4i;
typedef __attribute__((address_space(3))) v4i* lptr_v4i;
__device__ __forceinline__ gptr_v4i to_gbl(const void* p) {
  return (gptr_v4i)(unsigned long long)p;
}
__device__ __forceinline__ lptr_v4i to_lds(void* p) {
  // generic LDS address: low 32 bits are the LDS offset (ISA aperture rule)
  return (lptr_v4i)(unsigned)(unsigned long long)p;
}
#endif

__device__ __forceinline__ void wait_async0() {
#if USE_ASYNC_LDS
#if __has_builtin(__builtin_amdgcn_s_wait_asynccnt)
  __builtin_amdgcn_s_wait_asynccnt(0);
#else
  asm volatile("s_wait_asynccnt 0" ::: "memory");
#endif
#endif
}

// cumulative segment bounds (structural constants from the reference SEGS)
static const int BOUNDS[21] = {
    0, 16, 32, 5280, 6048, 57376, 73888, 90400, 106912, 141856, 158368,
    174880, 191392, 226336, 242848, 259360, 261424, 277936, 294448,
    310960, 311347};

__constant__ int dBOUNDS[21] = {
    0, 16, 32, 5280, 6048, 57376, 73888, 90400, 106912, 141856, 158368,
    174880, 191392, 226336, 242848, 259360, 261424, 277936, 294448,
    310960, 311347};

// Branchless guarded load of 8 consecutive floats along k (clamped index,
// unconditional load, cndmask zero).  Fast path for full-range chunks.
__device__ __forceinline__ void load_row8(const float* __restrict__ row,
                                          int k0, int kEnd, float v[8]) {
  if (k0 + 8 <= kEnd) {
#pragma unroll
    for (int u = 0; u < 8; ++u) v[u] = row[k0 + u];
  } else {
#pragma unroll
    for (int u = 0; u < 8; ++u) {
      const int k  = k0 + u;
      const int kc = (k < kEnd) ? k : (kEnd - 1);
      const float x = row[kc];
      v[u] = (k < kEnd) ? x : 0.0f;
    }
  }
}

// ---------------------------------------------------------------------------
// WMMA GEMM:  out[z][512,512] = op( A[z][:, Kstart:Kend] @ B + bias[z] )
//   A: row-major, element (m,k) at A[m*lda + k]   (k is ABSOLUTE)
//   B (!TRANSB): row-major [K,N], (k,n) at B[k*512 + n]   (k absolute)
//   B ( TRANSB): row-major [N,K], (k,n) at B[n*512 + k]   (X @ W^T)
// WG = 256 threads (8 waves), tile 64x64, K-chunk 32, double-buffered LDS,
// one barrier per chunk.  Wave w: rows 16*(w>>1), cols 32*(w&1).
// ---------------------------------------------------------------------------
template <bool RELU, bool TRANSB>
__global__ __launch_bounds__(256) void gemm512(
    const float* __restrict__ A, long lda, long aStrideZ,
    const float* __restrict__ B, const float* __restrict__ bias,
    int biasStrideZ, float* __restrict__ out, long outStrideZ, int Kstart,
    int Kend, int useBounds) {
  const int z = blockIdx.z;
  if (useBounds) {
    Kstart = dBOUNDS[z];
    Kend   = dBOUNDS[z + 1];
  }
  A    += (long)z * aStrideZ;
  bias += z * biasStrideZ;
  out  += (long)z * outStrideZ;

  // A staged as f32 (async-copy friendly, cvt at fragment read);
  // B staged pre-swizzled into the WMMA 16-bit fragment layout.
  __shared__ __align__(32) float  Af[2][64][32];      // 16 KB
  __shared__ __align__(32) __bf16 Bs[2][4][32][16];   // 8 KB

  const int tid   = threadIdx.x;
  const int lane  = tid & 31;
  const int w     = tid >> 5;
  const int waveM = w >> 1;  // 0..3
  const int waveN = w & 1;   // 0..1
  const int tileM = blockIdx.y * 64;
  const int tileN = blockIdx.x * 64;

  v8f c0 = {};
  v8f c1 = {};

  // A-writer: (row tid>>2, 8 consecutive k at (tid&3)*8) -> 32 B per thread
  const int amRow = tid >> 2;
  const int akCol = (tid & 3) * 8;
  const float* aRowP = A + (long)(tileM + amRow) * lda;

  float aV[8], bV[8];
  bool aInRegs = false;  // uniform: A chunk staged via registers (not async)

  auto fetchA = [&](int kb, int b) {
    if (kb + 32 <= Kend) {
#if USE_ASYNC_LDS
      const float* g = aRowP + kb + akCol;
      float* l = &Af[b][amRow][akCol];
      __builtin_amdgcn_global_load_async_to_lds_b128(to_gbl(g), to_lds(l), 0,
                                                     0);
      __builtin_amdgcn_global_load_async_to_lds_b128(to_gbl(g + 4),
                                                     to_lds(l + 4), 0, 0);
      aInRegs = false;
#else
#pragma unroll
      for (int u = 0; u < 8; ++u) aV[u] = aRowP[kb + akCol + u];
      aInRegs = true;
#endif
    } else {
      load_row8(aRowP, kb + akCol, Kend, aV);
      aInRegs = true;
    }
  };

  auto commitA = [&](int b) {
    if (aInRegs) {
      v4f lo = {aV[0], aV[1], aV[2], aV[3]};
      v4f hi = {aV[4], aV[5], aV[6], aV[7]};
      *(v4f*)&Af[b][amRow][akCol]     = lo;
      *(v4f*)&Af[b][amRow][akCol + 4] = hi;
    }
  };

  auto loadB = [&](int kb) {
    if (!TRANSB) {
      // one k per thread, 8 consecutive n (coalesced); clamp k branchlessly
      const int kl  = tid >> 3;
      const int nl0 = (tid & 7) * 8;
      const int k   = kb + kl;
      const bool ok = (k < Kend);
      const long kc = ok ? k : (Kend - 1);
      const float* br = B + kc * N_HID + (tileN + nl0);
#pragma unroll
      for (int u = 0; u < 8; ++u) {
        const float x = br[u];
        bV[u] = ok ? x : 0.0f;
      }
    } else {
      // one n per thread, 8 consecutive k (coalesced along W_ih rows)
      const int nl  = tid >> 2;
      const int kl0 = (tid & 3) * 8;
      load_row8(B + (long)(tileN + nl) * N_HID, kb + kl0, Kend, bV);
    }
  };

  auto storeB = [&](int b) {
    if (!TRANSB) {
      const int kl  = tid >> 3;
      const int nl0 = (tid & 7) * 8;
      const int j   = ((kl >> 4) << 3) + (kl & 7);
      const int lhi = ((kl >> 3) & 1) << 4;
#pragma unroll
      for (int u = 0; u < 8; ++u) {
        const int nl = nl0 + u;
        Bs[b][nl >> 4][(nl & 15) + lhi][j] = (__bf16)bV[u];
      }
    } else {
      const int nl  = tid >> 2;
      const int kl0 = (tid & 3) * 8;
      const int g   = nl >> 4;
      const int nlo = nl & 15;
#pragma unroll
      for (int u = 0; u < 8; ++u) {
        const int kl = kl0 + u;
        Bs[b][g][nlo + (((kl >> 3) & 1) << 4)][((kl >> 4) << 3) + (kl & 7)] =
            (__bf16)bV[u];
      }
    }
  };

  // -------- prologue: chunk 0 into buffer 0 --------
  fetchA(Kstart, 0);
  loadB(Kstart);
  commitA(0);
  storeB(0);
  wait_async0();

  int buf = 0;
  for (int kb = Kstart; kb < Kend; kb += 32) {
    __syncthreads();  // buffer `buf` ready; reads of buf^1 drained
    const int kn = kb + 32;
    const bool more = (kn < Kend);
    if (more) {  // issue next chunk's fetches under this chunk's WMMAs
      fetchA(kn, buf ^ 1);
      loadB(kn);
    }

    // ---- fragments ----
    // A 16-bit fragment: lanes 0-15 hold K {0..7,16..23}, lanes 16-31 +8.
    const float* afr = &Af[buf][waveM * 16 + (lane & 15)][(lane >> 4) * 8];
    const v8f f0 = *(const v8f*)afr;         // j = 0..7
    const v8f f1 = *(const v8f*)(afr + 16);  // j = 8..15
    v16bf a;
#pragma unroll
    for (int u = 0; u < 8; ++u) {
      a[u]     = (__bf16)f0[u];
      a[u + 8] = (__bf16)f1[u];
    }
    const v16bf b0 = *(const v16bf*)&Bs[buf][waveN * 2 + 0][lane][0];
    const v16bf b1 = *(const v16bf*)&Bs[buf][waveN * 2 + 1][lane][0];

    c0 = __builtin_amdgcn_wmma_f32_16x16x32_bf16(false, a, false, b0,
                                                 (short)0, c0, false, false);
    c1 = __builtin_amdgcn_wmma_f32_16x16x32_bf16(false, a, false, b1,
                                                 (short)0, c1, false, false);

    if (more) {
      commitA(buf ^ 1);
      storeB(buf ^ 1);
      if (kn + 32 < Kend) {  // global_prefetch_b8 two chunks ahead
        __builtin_prefetch(aRowP + kn + 32 + akCol, 0, 1);
        if (!TRANSB)
          __builtin_prefetch(B + (long)(kn + 32 + (tid >> 3)) * N_HID + tileN,
                             0, 1);
        else
          __builtin_prefetch(B + (long)(tileN + (tid >> 2)) * N_HID + kn + 32,
                             0, 1);
      }
    }
    wait_async0();  // next chunk's async copies must land before the barrier
    buf ^= 1;
  }

  // ---- epilogue: bias (+relu), f32 store ----
  // C/D layout: VGPR v, lanes 0-15 -> (M=v, N=lane); lanes 16-31 -> (M=v+8).
  const int mBase = tileM + waveM * 16 + ((lane >> 4) << 3);
  const int nBase = tileN + waveN * 32 + (lane & 15);
  const float bb0 = bias[nBase];
  const float bb1 = bias[nBase + 16];
#pragma unroll
  for (int v = 0; v < 8; ++v) {
    float x0 = c0[v] + bb0;
    float x1 = c1[v] + bb1;
    if (RELU) {
      x0 = fmaxf(x0, 0.0f);
      x1 = fmaxf(x1, 0.0f);
    }
    const long m = mBase + v;
    out[m * N_HID + nBase]      = x0;
    out[m * N_HID + nBase + 16] = x1;
  }
}

// ---------------------------------------------------------------------------
// prep: W_hh^T (so scan reads are coalesced), bias2 = b_ih + b_hh, h0=h1=0.
// ---------------------------------------------------------------------------
__global__ __launch_bounds__(256) void prep(
    const float* __restrict__ W_hh, const float* __restrict__ b_ih,
    const float* __restrict__ b_hh, float* __restrict__ Wt,
    float* __restrict__ bias2, float* __restrict__ h0,
    float* __restrict__ h1) {
  const int idx = blockIdx.x * 256 + threadIdx.x;
  if (idx < N_HID * N_HID) {
    const int j = idx & (N_HID - 1);
    const int k = idx >> 9;
    Wt[(long)k * N_HID + j] = W_hh[(long)j * N_HID + k];
  }
  if (idx < N_HID) {
    bias2[idx] = b_ih[idx] + b_hh[idx];
    h0[idx] = 0.0f;
    h1[idx] = 0.0f;
  }
}

// ---------------------------------------------------------------------------
// rnn_scan: h_t = tanh(xp_t + W_hh h_{t-1})   (xp already holds b_ih + b_hh)
// One WG of 512 threads (16 waves); h broadcast from LDS; Wt rows L2-resident.
// ---------------------------------------------------------------------------
__global__ __launch_bounds__(512) void rnn_scan(
    const float* __restrict__ xp, const float* __restrict__ Wt,
    float* __restrict__ ys, float* __restrict__ hstate) {
  __shared__ float hs[N_HID];
  const int j = threadIdx.x;
  hs[j] = hstate[j];
  __syncthreads();
  for (int t = 0; t < L_SEQ; ++t) {
    float acc = xp[(long)t * N_HID + j];
#pragma unroll 8
    for (int k = 0; k < N_HID; ++k)
      acc = fmaf(Wt[(long)k * N_HID + j], hs[k], acc);
    const float hn = tanhf(acc);
    ys[(long)t * N_HID + j] = hn;
    __syncthreads();
    hs[j] = hn;
    __syncthreads();
  }
  hstate[j] = hs[j];
}

// ---------------------------------------------------------------------------
// head: softmax(relu(out) @ W_out + b_out).  One wave32 per row.
// ---------------------------------------------------------------------------
__global__ __launch_bounds__(32) void head_softmax(
    const float* __restrict__ ys, const float* __restrict__ W_out,
    const float* __restrict__ b_out, float* __restrict__ out) {
  const int r = blockIdx.x;
  const int lane = threadIdx.x;
  float s0 = 0.0f, s1 = 0.0f, s2 = 0.0f;
  for (int j = lane; j < N_HID; j += 32) {
    const float v = fmaxf(ys[(long)r * N_HID + j], 0.0f);
    s0 = fmaf(v, W_out[j * 3 + 0], s0);
    s1 = fmaf(v, W_out[j * 3 + 1], s1);
    s2 = fmaf(v, W_out[j * 3 + 2], s2);
  }
#pragma unroll
  for (int off = 16; off >= 1; off >>= 1) {
    s0 += __shfl_xor(s0, off);
    s1 += __shfl_xor(s1, off);
    s2 += __shfl_xor(s2, off);
  }
  if (lane == 0) {
    const float x0 = s0 + b_out[0];
    const float x1 = s1 + b_out[1];
    const float x2 = s2 + b_out[2];
    const float m = fmaxf(x0, fmaxf(x1, x2));
    const float e0 = __expf(x0 - m);
    const float e1 = __expf(x1 - m);
    const float e2 = __expf(x2 - m);
    const float inv = 1.0f / (e0 + e1 + e2);
    out[r * 3 + 0] = e0 * inv;
    out[r * 3 + 1] = e1 * inv;
    out[r * 3 + 2] = e2 * inv;
  }
}

// ---------------------------------------------------------------------------
extern "C" void kernel_launch(void* const* d_in, const int* in_sizes, int n_in,
                              void* d_out, int out_size, void* d_ws,
                              size_t ws_size, hipStream_t stream) {
  (void)in_sizes; (void)n_in; (void)out_size;
  const float* input_list = (const float*)d_in[0];
  // d_in[1] = tensor_list (static segment bounds; hard-coded above)
  const float* W_cat    = (const float*)d_in[2];
  const float* b_branch = (const float*)d_in[3];
  const float* W_ih     = (const float*)d_in[4];
  const float* W_hh     = (const float*)d_in[5];
  const float* b_ih     = (const float*)d_in[6];
  const float* b_hh     = (const float*)d_in[7];
  const float* W_out    = (const float*)d_in[8];
  const float* b_out    = (const float*)d_in[9];
  float* out = (float*)d_out;

  const size_t MB  = 1u << 20;
  const long   MAT = (long)N_HID * N_HID;  // 262144 floats = 1 MB
  char* ws = (char*)d_ws;

  const dim3 gb(256);
  const dim3 g1(8, 8, 1);

  // ws_size is fixed per session -> deterministic schedule choice.
  if (ws_size >= 46 * MB) {
    // ---------------- batched schedule (~45 MB ws) ----------------
    float* Wt    = (float*)(ws);            // 1 MB  W_hh^T
    float* Y20   = (float*)(ws + 1 * MB);   // 20 MB branch activations
    float* XP20  = (float*)(ws + 21 * MB);  // 20 MB stage-1 projections
    float* O1    = (float*)(ws + 41 * MB);  // 1 MB
    float* O2    = (float*)(ws + 42 * MB);  // 1 MB
    float* XP2   = (float*)(ws + 43 * MB);  // 1 MB
    float* bias2 = (float*)(ws + 44 * MB);
    float* h0    = (float*)(ws + 44 * MB + 4096);
    float* h1    = (float*)(ws + 44 * MB + 8192);

    prep<<<dim3((N_HID * N_HID + 255) / 256), gb, 0, stream>>>(
        W_hh, b_ih, b_hh, Wt, bias2, h0, h1);

    // all 20 branch projections in one launch (1280 WGs, fills the GPU)
    gemm512<true, false><<<dim3(8, 8, 20), gb, 0, stream>>>(
        input_list, (long)F_TOTAL, 0L, W_cat, b_branch, N_HID, Y20, MAT, 0, 0,
        1);
    // all 20 stage-1 input projections in one launch
    gemm512<false, true><<<dim3(8, 8, 20), gb, 0, stream>>>(
        Y20, (long)N_HID, MAT, W_ih, bias2, 0, XP20, MAT, 0, N_HID, 0);

    for (int i = 0; i < 20; ++i) {
      rnn_scan<<<dim3(1), dim3(512), 0, stream>>>(XP20 + (long)i * MAT, Wt, O1,
                                                  h0);
      gemm512<false, true><<<g1, gb, 0, stream>>>(O1, (long)N_HID, 0L, W_ih,
                                                  bias2, 0, XP2, 0L, 0, N_HID,
                                                  0);
      rnn_scan<<<dim3(1), dim3(512), 0, stream>>>(XP2, Wt, O2, h1);
    }
    head_softmax<<<dim3(L_SEQ), dim3(32), 0, stream>>>(O2, W_out, b_out, out);
  } else {
    // ---------------- compact per-branch schedule (~5.3 MB ws) ----------------
    float* Wt    = (float*)(ws);           // 1 MB
    float* Y     = (float*)(ws + 1 * MB);  // 1 MB
    float* XP    = (float*)(ws + 2 * MB);  // 1 MB
    float* O1    = (float*)(ws + 3 * MB);  // 1 MB
    float* O2    = (float*)(ws + 4 * MB);  // 1 MB
    float* bias2 = (float*)(ws + 5 * MB);
    float* h0    = (float*)(ws + 5 * MB + 4096);
    float* h1    = (float*)(ws + 5 * MB + 8192);

    prep<<<dim3((N_HID * N_HID + 255) / 256), gb, 0, stream>>>(
        W_hh, b_ih, b_hh, Wt, bias2, h0, h1);

    for (int i = 0; i < 20; ++i) {
      gemm512<true, false><<<g1, gb, 0, stream>>>(
          input_list, (long)F_TOTAL, 0L, W_cat, b_branch + i * N_HID, 0, Y, 0L,
          BOUNDS[i], BOUNDS[i + 1], 0);
      gemm512<false, true><<<g1, gb, 0, stream>>>(Y, (long)N_HID, 0L, W_ih,
                                                  bias2, 0, XP, 0L, 0, N_HID,
                                                  0);
      rnn_scan<<<dim3(1), dim3(512), 0, stream>>>(XP, Wt, O1, h0);
      gemm512<false, true><<<g1, gb, 0, stream>>>(O1, (long)N_HID, 0L, W_ih,
                                                  bias2, 0, XP, 0L, 0, N_HID,
                                                  0);
      rnn_scan<<<dim3(1), dim3(512), 0, stream>>>(XP, Wt, O2, h1);
    }
    head_softmax<<<dim3(L_SEQ), dim3(32), 0, stream>>>(O2, W_out, b_out, out);
  }
}
